// MambaTower_27161373179955
// MI455X (gfx1250) — matrix operand order repaired and checked by reading, not compiled, not run on hardware
//
#include <hip/hip_runtime.h>
#include <math.h>

// ---------------- problem constants ----------------
#define D_MODEL   768
#define SEQ       4096
#define BATCH     2
#define BT        (BATCH*SEQ)        // 8192 tokens
#define D_STATE   16
#define D_CONV    4
#define D_INNER   1536
#define HEADDIM   64
#define NHEADS    24
#define CONV_DIM  (D_INNER + 2*D_STATE)            // 1568
#define D_IN_PROJ (2*D_INNER + 2*D_STATE + NHEADS) // 3128
#define CHUNK     64
#define NCHUNK    (SEQ/CHUNK)        // 64
#define EPSF      1e-5f

typedef __attribute__((ext_vector_type(16))) __bf16 v16bf;
typedef __attribute__((ext_vector_type(8)))  float  v8f;

// =====================================================================
// GEMM: C[M,N] = A[M,K] (fp32) * W[K,N] (fp32), bf16 WMMA, fp32 accum.
// Software-pipelined: global loads for K-tile k+1 are issued in named
// registers before the WMMA compute of tile k, so loadcnt waits overlap
// a full compute phase (no per-load s_wait_loadcnt serialization).
// Block: 256 thr = 8 waves, tile 128x128, BK=32. Wave grid 2(M) x 4(N).
// =====================================================================
__global__ __launch_bounds__(256) void gemm_bf16_wmma(
    const float* __restrict__ A, const float* __restrict__ W,
    const float* __restrict__ res, float* __restrict__ C,
    int M, int N, int K)
{
  __shared__ __bf16 As[128][36];   // [m][k], padded stride 36 (anti-bank-conflict)
  __shared__ __bf16 Bs[128][36];   // [n][k] (transposed on store)

  const int tid   = threadIdx.x;
  const int lane  = tid & 31;
  const int wave  = tid >> 5;
  const int waveM = wave >> 2;     // 0..1
  const int waveN = wave & 3;      // 0..3
  const int m0 = blockIdx.y * 128;
  const int n0 = blockIdx.x * 128;
  const bool fullN = (n0 + 128 <= N);   // uniform per block

  // Per-thread staging slice coordinates (constant across K-tiles).
  int aRow[4], aCol[4], wKr[4], wNv[4];
  #pragma unroll
  for (int i = 0; i < 4; ++i) {
    int q   = tid + 256*i;
    aRow[i] = q >> 3;              // 0..127
    aCol[i] = (q & 7) << 2;        // 0,4,...,28
    wKr[i]  = q >> 5;              // 0..31
    wNv[i]  = (q & 31) << 2;       // 0..124
  }

  float4 aReg[4], wReg[4];

  auto loadTiles = [&](int kk) {
    #pragma unroll
    for (int i = 0; i < 4; ++i)
      aReg[i] = *reinterpret_cast<const float4*>(
          A + (size_t)(m0 + aRow[i])*K + kk + aCol[i]);
    if (fullN) {
      #pragma unroll
      for (int i = 0; i < 4; ++i)
        wReg[i] = *reinterpret_cast<const float4*>(
            W + (size_t)(kk + wKr[i])*N + n0 + wNv[i]);
    } else {
      #pragma unroll
      for (int i = 0; i < 4; ++i) {
        int col = n0 + wNv[i];
        wReg[i] = make_float4(0.f, 0.f, 0.f, 0.f);
        if (col < N)               // N % 4 == 0: float4 never straddles edge
          wReg[i] = *reinterpret_cast<const float4*>(
              W + (size_t)(kk + wKr[i])*N + col);
      }
    }
  };

  auto storeTiles = [&]() {
    #pragma unroll
    for (int i = 0; i < 4; ++i) {
      As[aRow[i]][aCol[i]+0] = (__bf16)aReg[i].x;
      As[aRow[i]][aCol[i]+1] = (__bf16)aReg[i].y;
      As[aRow[i]][aCol[i]+2] = (__bf16)aReg[i].z;
      As[aRow[i]][aCol[i]+3] = (__bf16)aReg[i].w;
      Bs[wNv[i]+0][wKr[i]] = (__bf16)wReg[i].x;
      Bs[wNv[i]+1][wKr[i]] = (__bf16)wReg[i].y;
      Bs[wNv[i]+2][wKr[i]] = (__bf16)wReg[i].z;
      Bs[wNv[i]+3][wKr[i]] = (__bf16)wReg[i].w;
    }
  };

  v8f acc[4][2] = {};

  loadTiles(0);
  for (int kk = 0; kk < K; kk += 32) {
    __syncthreads();               // previous compute done; LDS free
    storeTiles();
    __syncthreads();
    if (kk + 32 < K) loadTiles(kk + 32);   // prefetch next tile into regs

    // ---- B fragments (32x16, 16-bit): lanes0-15 K=0..15, lanes16-31 K=16..31 ----
    v16bf bfr[2];
    #pragma unroll
    for (int nt = 0; nt < 2; ++nt) {
      const int ncol = 32*waveN + 16*nt + (lane & 15);
      const int kb   = (lane & 16) ? 16 : 0;
      #pragma unroll
      for (int e = 0; e < 16; ++e) bfr[nt][e] = Bs[ncol][kb + e];
    }
    // ---- A fragments (16x32, 16-bit) + WMMA ----
    #pragma unroll
    for (int mt = 0; mt < 4; ++mt) {
      const int r  = 64*waveM + 16*mt + (lane & 15);
      const int kq = (lane & 16) ? 16 : 0;  // lanes16-31 hold K 16..31
      v16bf af;
      #pragma unroll
      for (int j = 0; j < 4; ++j) {
        af[2*j]     = As[r][kq + 2*j];
        af[2*j+1]   = As[r][kq + 2*j + 1];
        af[8+2*j]   = As[r][8 + kq + 2*j];
        af[8+2*j+1] = As[r][8 + kq + 2*j + 1];
      }
      #pragma unroll
      for (int nt = 0; nt < 2; ++nt)
        acc[mt][nt] = __builtin_amdgcn_wmma_f32_16x16x32_bf16(
            false, af, false, bfr[nt], (short)0, acc[mt][nt], false, false);
    }
  }

  // ---- store D (C/D layout: VGPR j -> M=j (+8 for hi lanes), N=lane%16) ----
  #pragma unroll
  for (int mt = 0; mt < 4; ++mt) {
    #pragma unroll
    for (int nt = 0; nt < 2; ++nt) {
      const int n = n0 + 32*waveN + 16*nt + (lane & 15);
      if (n >= N) continue;
      const int mb = m0 + 64*waveM + 16*mt + ((lane & 16) ? 8 : 0);
      #pragma unroll
      for (int j = 0; j < 8; ++j) {
        const size_t idx = (size_t)(mb + j) * N + n;
        float v = acc[mt][nt][j];
        if (res) v += res[idx];
        C[idx] = v;
      }
    }
  }
}

// =====================================================================
// Depthwise causal conv1d (width 4) + bias + SiLU over xBC channels.
// =====================================================================
__global__ __launch_bounds__(256) void conv_silu_kernel(
    const float* __restrict__ zx, const float* __restrict__ cw,
    const float* __restrict__ cb, float* __restrict__ xBC)
{
  int gid = blockIdx.x * blockDim.x + threadIdx.x;
  if (gid >= BT * CONV_DIM) return;
  int c   = gid % CONV_DIM;
  int row = gid / CONV_DIM;
  int t   = row & (SEQ - 1);
  float acc = cb[c];
  #pragma unroll
  for (int k = 0; k < D_CONV; ++k) {
    int tt = t - (D_CONV - 1) + k;
    if (tt >= 0)
      acc += zx[(size_t)(row - (D_CONV-1) + k)*D_IN_PROJ + D_INNER + c]
             * cw[c*D_CONV + k];
  }
  xBC[(size_t)row*CONV_DIM + c] = acc / (1.f + __expf(-acc));  // silu
}

// =====================================================================
// dt = softplus(dt_raw + dt_bias); dA = exp(dt * -exp(A_log))
// =====================================================================
__global__ __launch_bounds__(256) void dt_kernel(
    const float* __restrict__ zx, const float* __restrict__ dtb,
    const float* __restrict__ Alog,
    float* __restrict__ dt_o, float* __restrict__ dA_o)
{
  int gid = blockIdx.x * blockDim.x + threadIdx.x;
  if (gid >= BT * NHEADS) return;
  int h   = gid % NHEADS;
  int row = gid / NHEADS;
  float x  = zx[(size_t)row*D_IN_PROJ + (D_INNER + CONV_DIM) + h] + dtb[h];
  float dt = (x > 20.f) ? x : log1pf(__expf(x));
  dt_o[gid] = dt;
  dA_o[gid] = __expf(dt * (-__expf(Alog[h])));
}

// =====================================================================
// Scan pass 1: per-(b,h,chunk) wave computes chunk state (zero init)
// and the product of dA over the chunk. Lane l owns rows p=2l,2l+1 of
// the 64x16 state (32 regs).
// =====================================================================
__global__ __launch_bounds__(128) void scan_state_kernel(
    const float* __restrict__ xBC, const float* __restrict__ dtv,
    const float* __restrict__ dAv, float* __restrict__ S, float* __restrict__ P)
{
  const int wid   = (blockIdx.x * blockDim.x + threadIdx.x) >> 5;
  const int lane  = threadIdx.x & 31;
  const int chunk = wid & (NCHUNK - 1);
  const int h     = (wid >> 6) % NHEADS;
  const int b     = wid / (NCHUNK * NHEADS);
  const int hcol  = h * HEADDIM;
  const int p0    = 2*lane;

  float h0[D_STATE], h1[D_STATE];
  #pragma unroll
  for (int n = 0; n < D_STATE; ++n) { h0[n] = 0.f; h1[n] = 0.f; }
  float prod = 1.f;

  const int rowBase = b*SEQ + chunk*CHUNK;
  for (int s = 0; s < CHUNK; ++s) {
    const int row = rowBase + s;
    const float* xr = xBC + (size_t)row * CONV_DIM;
    const float x0 = xr[hcol + p0];
    const float x1 = xr[hcol + p0 + 1];
    float Bv[D_STATE];
    #pragma unroll
    for (int q = 0; q < 4; ++q) {
      float4 v = *reinterpret_cast<const float4*>(xr + D_INNER + 4*q);
      Bv[4*q] = v.x; Bv[4*q+1] = v.y; Bv[4*q+2] = v.z; Bv[4*q+3] = v.w;
    }
    const float dt = dtv[row*NHEADS + h];
    const float dA = dAv[row*NHEADS + h];
    prod *= dA;
    const float c0 = dt * x0, c1 = dt * x1;
    #pragma unroll
    for (int n = 0; n < D_STATE; ++n) {
      h0[n] = fmaf(h0[n], dA, c0 * Bv[n]);
      h1[n] = fmaf(h1[n], dA, c1 * Bv[n]);
    }
  }
  float* Sp = S + (size_t)wid * (HEADDIM * D_STATE);
  #pragma unroll
  for (int n = 0; n < D_STATE; ++n) {
    Sp[p0*D_STATE + n]     = h0[n];
    Sp[(p0+1)*D_STATE + n] = h1[n];
  }
  if (lane == 0) P[wid] = prod;
}

// =====================================================================
// Scan pass 2: sequential cross-chunk recombine (48 waves, 64 steps).
// Writes per-chunk initial state: Hinit[c] = h; h = h*P[c] + S[c].
// =====================================================================
__global__ __launch_bounds__(32) void scan_combine_kernel(
    const float* __restrict__ S, const float* __restrict__ P,
    float* __restrict__ Hinit)
{
  const int wid  = blockIdx.x;       // b*NHEADS + h
  const int lane = threadIdx.x;
  const int p0   = 2*lane;
  float h0[D_STATE], h1[D_STATE];
  #pragma unroll
  for (int n = 0; n < D_STATE; ++n) { h0[n] = 0.f; h1[n] = 0.f; }

  for (int c = 0; c < NCHUNK; ++c) {
    const size_t base = (size_t)(wid*NCHUNK + c) * (HEADDIM * D_STATE);
    float* Hp = Hinit + base;
    const float* Sp = S + base;
    const float p = P[wid*NCHUNK + c];
    #pragma unroll
    for (int n = 0; n < D_STATE; ++n) {
      Hp[p0*D_STATE + n]     = h0[n];
      Hp[(p0+1)*D_STATE + n] = h1[n];
      h0[n] = fmaf(h0[n], p, Sp[p0*D_STATE + n]);
      h1[n] = fmaf(h1[n], p, Sp[(p0+1)*D_STATE + n]);
    }
  }
}

// =====================================================================
// Scan pass 3: per-chunk output recompute with correct initial state.
// y[t,p] = sum_n h[p][n]*C[n] + D_skip*xh   (h updated before dot).
// =====================================================================
__global__ __launch_bounds__(128) void scan_output_kernel(
    const float* __restrict__ xBC, const float* __restrict__ dtv,
    const float* __restrict__ dAv, const float* __restrict__ Hinit,
    const float* __restrict__ Dsk, float* __restrict__ y)
{
  const int wid   = (blockIdx.x * blockDim.x + threadIdx.x) >> 5;
  const int lane  = threadIdx.x & 31;
  const int chunk = wid & (NCHUNK - 1);
  const int h     = (wid >> 6) % NHEADS;
  const int b     = wid / (NCHUNK * NHEADS);
  const int hcol  = h * HEADDIM;
  const int p0    = 2*lane;

  const float* Hp = Hinit + (size_t)wid * (HEADDIM * D_STATE);
  float h0[D_STATE], h1[D_STATE];
  #pragma unroll
  for (int n = 0; n < D_STATE; ++n) {
    h0[n] = Hp[p0*D_STATE + n];
    h1[n] = Hp[(p0+1)*D_STATE + n];
  }
  const float dskip = Dsk[h];

  const int rowBase = b*SEQ + chunk*CHUNK;
  for (int s = 0; s < CHUNK; ++s) {
    const int row = rowBase + s;
    const float* xr = xBC + (size_t)row * CONV_DIM;
    const float x0 = xr[hcol + p0];
    const float x1 = xr[hcol + p0 + 1];
    float Bv[D_STATE], Cv[D_STATE];
    #pragma unroll
    for (int q = 0; q < 4; ++q) {
      float4 vb = *reinterpret_cast<const float4*>(xr + D_INNER + 4*q);
      float4 vc = *reinterpret_cast<const float4*>(xr + D_INNER + D_STATE + 4*q);
      Bv[4*q] = vb.x; Bv[4*q+1] = vb.y; Bv[4*q+2] = vb.z; Bv[4*q+3] = vb.w;
      Cv[4*q] = vc.x; Cv[4*q+1] = vc.y; Cv[4*q+2] = vc.z; Cv[4*q+3] = vc.w;
    }
    const float dt = dtv[row*NHEADS + h];
    const float dA = dAv[row*NHEADS + h];
    const float c0 = dt * x0, c1 = dt * x1;
    float y0 = 0.f, y1 = 0.f;
    #pragma unroll
    for (int n = 0; n < D_STATE; ++n) {
      h0[n] = fmaf(h0[n], dA, c0 * Bv[n]);
      h1[n] = fmaf(h1[n], dA, c1 * Bv[n]);
      y0 += h0[n] * Cv[n];
      y1 += h1[n] * Cv[n];
    }
    y[(size_t)row*D_INNER + hcol + p0]     = fmaf(dskip, x0, y0);
    y[(size_t)row*D_INNER + hcol + p0 + 1] = fmaf(dskip, x1, y1);
  }
}

// =====================================================================
// y *= silu(z); RMSNorm over 1536 with weight gw.
// =====================================================================
__global__ __launch_bounds__(256) void gate_rmsnorm_kernel(
    const float* __restrict__ y, const float* __restrict__ zx,
    const float* __restrict__ gw, float* __restrict__ out)
{
  __shared__ float red[256];
  const int row = blockIdx.x;
  const int tid = threadIdx.x;
  float v[6]; float ss = 0.f;
  #pragma unroll
  for (int i = 0; i < 6; ++i) {
    int c = tid + 256*i;
    float yv = y[(size_t)row*D_INNER + c];
    float zv = zx[(size_t)row*D_IN_PROJ + c];
    float g = yv * (zv / (1.f + __expf(-zv)));
    v[i] = g; ss += g*g;
  }
  red[tid] = ss; __syncthreads();
  for (int s = 128; s > 0; s >>= 1) {
    if (tid < s) red[tid] += red[tid + s];
    __syncthreads();
  }
  const float r = rsqrtf(red[0] / (float)D_INNER + EPSF);
  #pragma unroll
  for (int i = 0; i < 6; ++i) {
    int c = tid + 256*i;
    out[(size_t)row*D_INNER + c] = v[i] * r * gw[c];
  }
}

// =====================================================================
// LayerNorm over 768 (input already has residual added by GEMM).
// =====================================================================
__global__ __launch_bounds__(256) void layernorm_kernel(
    const float* __restrict__ inp, const float* __restrict__ lw,
    const float* __restrict__ lb, float* __restrict__ out)
{
  __shared__ float red[256];
  const int row = blockIdx.x;
  const int tid = threadIdx.x;
  float v[3]; float s = 0.f;
  #pragma unroll
  for (int i = 0; i < 3; ++i) {
    int c = tid + 256*i;
    v[i] = inp[(size_t)row*D_MODEL + c];
    s += v[i];
  }
  red[tid] = s; __syncthreads();
  for (int st = 128; st > 0; st >>= 1) {
    if (tid < st) red[tid] += red[tid + st];
    __syncthreads();
  }
  const float mu = red[0] / (float)D_MODEL;
  __syncthreads();
  float s2 = 0.f;
  #pragma unroll
  for (int i = 0; i < 3; ++i) { float d = v[i] - mu; s2 += d*d; }
  red[tid] = s2; __syncthreads();
  for (int st = 128; st > 0; st >>= 1) {
    if (tid < st) red[tid] += red[tid + st];
    __syncthreads();
  }
  const float r = rsqrtf(red[0] / (float)D_MODEL + EPSF);
  #pragma unroll
  for (int i = 0; i < 3; ++i) {
    int c = tid + 256*i;
    out[(size_t)row*D_MODEL + c] = (v[i] - mu) * r * lw[c] + lb[c];
  }
}

// =====================================================================
// Host orchestration. Workspace need: ~332 MB (all fp32 scratch).
// =====================================================================
extern "C" void kernel_launch(void* const* d_in, const int* in_sizes, int n_in,
                              void* d_out, int out_size, void* d_ws, size_t ws_size,
                              hipStream_t stream)
{
  (void)in_sizes; (void)n_in; (void)out_size; (void)ws_size;
  const float* x0      = (const float*)d_in[0];
  const float* in_proj = (const float*)d_in[1];
  const float* conv_w  = (const float*)d_in[2];
  const float* conv_b  = (const float*)d_in[3];
  const float* dt_bias = (const float*)d_in[4];
  const float* A_log   = (const float*)d_in[5];
  const float* D_skip  = (const float*)d_in[6];
  const float* gnorm_w = (const float*)d_in[7];
  const float* out_prj = (const float*)d_in[8];
  const float* ln_w    = (const float*)d_in[9];
  const float* ln_b    = (const float*)d_in[10];

  float* ws    = (float*)d_ws;
  float* zx    = ws;                                              // BT*3128
  float* xbc   = zx    + (size_t)BT*D_IN_PROJ;                    // BT*1568
  float* dtbuf = xbc   + (size_t)BT*CONV_DIM;                     // BT*24
  float* dAbuf = dtbuf + (size_t)BT*NHEADS;                       // BT*24
  float* Sbuf  = dAbuf + (size_t)BT*NHEADS;                       // 48*64*1024
  float* Pbuf  = Sbuf  + (size_t)BATCH*NHEADS*NCHUNK*HEADDIM*D_STATE; // 3072
  float* Hbuf  = Pbuf  + (size_t)BATCH*NHEADS*NCHUNK;             // 48*64*1024
  float* ybuf  = Hbuf  + (size_t)BATCH*NHEADS*NCHUNK*HEADDIM*D_STATE; // BT*1536
  float* gbuf  = ybuf  + (size_t)BT*D_INNER;                      // BT*1536
  float* pre   = gbuf  + (size_t)BT*D_INNER;                      // BT*768
  float* xmid  = pre   + (size_t)BT*D_MODEL;                      // BT*768

  const int scanBlocks = (BATCH*NHEADS*NCHUNK*32) / 128;          // 768

  for (int l = 0; l < 2; ++l) {
    const float* xin  = (l == 0) ? x0 : xmid;
    float*       xout = (l == 0) ? xmid : (float*)d_out;
    const float* Wi  = in_proj + (size_t)l*D_MODEL*D_IN_PROJ;
    const float* cw  = conv_w  + (size_t)l*CONV_DIM*D_CONV;
    const float* cb  = conv_b  + (size_t)l*CONV_DIM;
    const float* dtb = dt_bias + (size_t)l*NHEADS;
    const float* Al  = A_log   + (size_t)l*NHEADS;
    const float* Dl  = D_skip  + (size_t)l*NHEADS;
    const float* gw  = gnorm_w + (size_t)l*D_INNER;
    const float* Wo  = out_prj + (size_t)l*D_INNER*D_MODEL;
    const float* lw  = ln_w    + (size_t)l*D_MODEL;
    const float* lb  = ln_b    + (size_t)l*D_MODEL;

    dim3 g1((D_IN_PROJ + 127)/128, BT/128);
    gemm_bf16_wmma<<<g1, 256, 0, stream>>>(xin, Wi, nullptr, zx,
                                           BT, D_IN_PROJ, D_MODEL);
    conv_silu_kernel<<<(BT*CONV_DIM + 255)/256, 256, 0, stream>>>(zx, cw, cb, xbc);
    dt_kernel<<<(BT*NHEADS + 255)/256, 256, 0, stream>>>(zx, dtb, Al, dtbuf, dAbuf);
    scan_state_kernel<<<scanBlocks, 128, 0, stream>>>(xbc, dtbuf, dAbuf, Sbuf, Pbuf);
    scan_combine_kernel<<<BATCH*NHEADS, 32, 0, stream>>>(Sbuf, Pbuf, Hbuf);
    scan_output_kernel<<<scanBlocks, 128, 0, stream>>>(xbc, dtbuf, dAbuf, Hbuf, Dl, ybuf);
    gate_rmsnorm_kernel<<<BT, 256, 0, stream>>>(ybuf, zx, gw, gbuf);
    dim3 g2(D_MODEL/128, BT/128);
    gemm_bf16_wmma<<<g2, 256, 0, stream>>>(gbuf, Wo, xin, pre,
                                           BT, D_MODEL, D_INNER);
    layernorm_kernel<<<BT, 256, 0, stream>>>(pre, lw, lb, xout);
  }
}
